// EquiAdapter_10273561772109
// MI455X (gfx1250) — matrix operand order repaired and checked by reading, not compiled, not run on hardware
//
#include <hip/hip_runtime.h>
#include <hip/hip_bf16.h>

// ---------------------------------------------------------------------------
// EquiAdapter fused kernel for MI455X (gfx1250), bf16 WMMA / f32 accumulate.
// ---------------------------------------------------------------------------

typedef __attribute__((ext_vector_type(16))) __bf16 v16bf;
typedef __attribute__((ext_vector_type(8)))  __bf16 v8bf;
typedef __attribute__((ext_vector_type(8)))  float  v8f;

#define WMMA_BF16(A, B, C) \
    __builtin_amdgcn_wmma_f32_16x16x32_bf16(false, (A), false, (B), (short)0, (C), false, false)

// Problem constants
#define NB   8      // batch
#define NE   4      // n_equi
#define NT   1024   // tokens
#define DEQ  1024   // d_eq
#define NG   8      // n_group
#define BK   128    // blocks
#define TT   64     // text tokens
#define DL   2048   // d_llm
#define SCALE 0.08838834764831845f   // 128^-0.5

// Workspace byte offsets
#define WS_WE   0
#define WS_WQ   32768
#define WS_WG   65536
#define WS_WB   98304
#define WS_K    131072   // k  bf16  [B][64][128]
#define WS_VT   262144   // vT bf16  [B][128][64]

// ---------------------------------------------------------------------------
// A operand: 16x32 bf16 tile, rows m = lane&15, K contiguous (matches B):
//   elems 0..15 -> K = k0 + (lane>=16 ? 16 : 0) + e
// base points at row 0 of the M-tile; stride in elements; 16B-aligned chunks.
// ---------------------------------------------------------------------------
__device__ __forceinline__ v16bf load_a_tile(const __bf16* base, int stride,
                                             int k0, int l15, int half) {
    const __bf16* p = base + l15 * stride + k0 + half * 16;
    v8bf lo = *(const v8bf*)(p);
    v8bf hi = *(const v8bf*)(p + 8);
    v16bf a;
#pragma unroll
    for (int i = 0; i < 8; ++i) { a[i] = lo[i]; a[i + 8] = hi[i]; }
    return a;
}

// ---------------------------------------------------------------------------
// B operand: 32x16 bf16 tile (K x N). Lane holds column N = lane&15; this code's K order (as A):
// K = k0 + (lane>=16 ? 16 : 0) + e. We stage B transposed ([N][K]-major, i.e.
// exactly the natural W[o][i] weight layout), so each lane reads 16 contiguous
// bf16 (32 B).
// ---------------------------------------------------------------------------
__device__ __forceinline__ v16bf load_b_tile(const __bf16* Wt, int ld,
                                             int n0, int k0, int l15, int half) {
    const __bf16* p = Wt + (size_t)(n0 + l15) * ld + k0 + half * 16;
    v8bf lo = *(const v8bf*)(p);
    v8bf hi = *(const v8bf*)(p + 8);
    v16bf b;
#pragma unroll
    for (int i = 0; i < 8; ++i) { b[i] = lo[i]; b[i + 8] = hi[i]; }
    return b;
}

// C/D layout: lane -> N = lane&15 ; VGPR r -> M = r + (lane>=16 ? 8 : 0)
__device__ __forceinline__ void store_c_bf(__bf16* buf, int stride, int n0,
                                           const v8f acc, int l15, int half) {
#pragma unroll
    for (int r = 0; r < 8; ++r)
        buf[(r + half * 8) * stride + n0 + l15] = (__bf16)acc[r];
}

// ---------------------------------------------------------------------------
// Kernel 0: convert the four 128x128 f32 weight matrices to bf16 workspace.
// ---------------------------------------------------------------------------
__global__ __launch_bounds__(256)
void equi_prep_weights(const float* __restrict__ Wq, const float* __restrict__ Wg,
                       const float* __restrict__ Wb, const float* __restrict__ We,
                       __bf16* __restrict__ wq_bf, __bf16* __restrict__ wg_bf,
                       __bf16* __restrict__ wb_bf, __bf16* __restrict__ we_bf) {
    int idx = blockIdx.x * 256 + threadIdx.x;
    if (idx < BK * BK) {
        wq_bf[idx] = (__bf16)Wq[idx];
        wg_bf[idx] = (__bf16)Wg[idx];
        wb_bf[idx] = (__bf16)Wb[idx];
        we_bf[idx] = (__bf16)We[idx];
    }
}

// ---------------------------------------------------------------------------
// Kernel 1: k = h_llm@Wk^T+bk, v = h_llm@Wv^T+bv  (tiny, L2 resident).
// Stores k bf16 as [s][o] (B operand of q.k^T) and v bf16 transposed [o][s]
// (B operand of attn.v).
// ---------------------------------------------------------------------------
__global__ __launch_bounds__(128)
void equi_kv_kernel(const float* __restrict__ h_llm,
                    const float* __restrict__ Wk, const float* __restrict__ bk,
                    const float* __restrict__ Wv, const float* __restrict__ bv,
                    __bf16* __restrict__ k_bf, __bf16* __restrict__ vT_bf) {
    __shared__ float row[DL];
    const int b = blockIdx.x >> 6;
    const int s = blockIdx.x & 63;
    const float* src = h_llm + ((size_t)b * TT + s) * DL;
    for (int i = threadIdx.x; i < DL; i += 128) row[i] = src[i];
    __syncthreads();

    const int o = threadIdx.x;  // 128 outputs
    const float* wkr = Wk + (size_t)o * DL;
    const float* wvr = Wv + (size_t)o * DL;
    float kacc = bk[o], vacc = bv[o];
#pragma unroll 4
    for (int i = 0; i < DL; ++i) {
        float x = row[i];
        kacc = fmaf(x, wkr[i], kacc);
        vacc = fmaf(x, wvr[i], vacc);
    }
    k_bf[((size_t)b * TT + s) * BK + o]  = (__bf16)kacc;
    vT_bf[((size_t)b * BK + o) * TT + s] = (__bf16)vacc;
}

// ---------------------------------------------------------------------------
// Kernel 2: fused main kernel. One block = 16 tokens of one (b, n_equi).
// 256 threads = 8 wave32s. 2048 blocks total.
// ---------------------------------------------------------------------------
#define HS   1032   // h_bf row stride (1024 + 8), 16B-aligned rows
#define S136 136    // 128-wide bf16 buffers
#define S72  72     // attn stride (64 + 8)
#define S68  68     // scores f32 stride

__global__ __launch_bounds__(256)
void equi_main_kernel(const float* __restrict__ h_prime,
                      const float* __restrict__ bq, const float* __restrict__ bg,
                      const float* __restrict__ bb,
                      const __bf16* __restrict__ we_bf, const __bf16* __restrict__ wq_bf,
                      const __bf16* __restrict__ wg_bf, const __bf16* __restrict__ wb_bf,
                      const __bf16* __restrict__ k_bf_g, const __bf16* __restrict__ vT_bf_g,
                      float* __restrict__ out) {
    __shared__ __align__(16) unsigned char smem[61440];
    __bf16* h_bf     = (__bf16*)(smem);                    // 16 x 1032     33024 B
    __bf16* inv_bf   = (__bf16*)(smem + 33024);            // 16 x 136       4352 B
    __bf16* q_bf     = (__bf16*)(smem + 37376);            // 16 x 136       4352 B
    __bf16* attn_bf  = (__bf16*)(smem + 41728);            // 16 x 72        2304 B
    __bf16* ctx_bf   = (__bf16*)(smem + 44032);            // 16 x 136       4352 B
    __bf16* gamma_bf = (__bf16*)(smem + 48384);            // 16 x 136       4352 B
    __bf16* beta_bf  = (__bf16*)(smem + 52736);            // 16 x 136       4352 B
    float*  scores_f = (float*) (smem + 57088);            // 16 x 68        4352 B
    __bf16* gate_bf  = (__bf16*)(smem + 57088);            // reuse after softmax

    const int tid  = threadIdx.x;
    const int lane = tid & 31;
    const int wv   = tid >> 5;       // wave id 0..7
    const int half = lane >> 4;      // 0/1
    const int l15  = lane & 15;

    const int blk = blockIdx.x;           // 0..2047
    const int b   = blk >> 8;             // batch
    const int ne  = (blk >> 6) & 3;       // equi index
    const int t0  = (blk & 63) << 4;      // token tile base

    const float* hp_t = h_prime + (((size_t)b * NE + ne) * NT + t0) * DEQ;
    float*       op_t = out     + (((size_t)b * NE + ne) * NT + t0) * DEQ;
    const __bf16* k_b  = k_bf_g  + (size_t)b * TT * BK;
    const __bf16* vT_b = vT_bf_g + (size_t)b * BK * TT;

    // ---- Phase 0: load f32 tile -> bf16 LDS + fused group-max pooling ----
    {
        const int t = tid >> 4;          // token row 0..15
        const int c0 = tid & 15;         // base block column
#pragma unroll
        for (int p = 0; p < 8; ++p) {
            const int c = c0 + p * 16;   // block index 0..127
            float mx = -3.4e38f;
#pragma unroll
            for (int g = 0; g < NG; ++g) {
                float x = hp_t[(size_t)t * DEQ + g * BK + c];
                h_bf[t * HS + g * BK + c] = (__bf16)x;
                mx = fmaxf(mx, x);
            }
            inv_bf[t * S136 + c] = (__bf16)mx;
        }
    }
    __syncthreads();

    // ---- Phase 1: q = inv @ Wq^T + bq  (wave w -> N-tile w) ----
    {
        const int n0 = wv * 16;
        const float bias = bq[n0 + l15];
        v8f acc;
#pragma unroll
        for (int r = 0; r < 8; ++r) acc[r] = bias;
#pragma unroll
        for (int ks = 0; ks < 4; ++ks) {
            v16bf a  = load_a_tile(inv_bf, S136, ks * 32, l15, half);
            v16bf bm = load_b_tile(wq_bf, BK, n0, ks * 32, l15, half);
            acc = WMMA_BF16(a, bm, acc);
        }
        store_c_bf(q_bf, S136, n0, acc, l15, half);
    }
    __syncthreads();

    // ---- Phase 2: scores = (q @ k^T) * SCALE  (waves 0..3, N = text) ----
    if (wv < 4) {
        const int s0 = wv * 16;
        v8f acc = {};
#pragma unroll
        for (int ks = 0; ks < 4; ++ks) {
            v16bf a  = load_a_tile(q_bf, S136, ks * 32, l15, half);
            v16bf bm = load_b_tile(k_b, BK, s0, ks * 32, l15, half);
            acc = WMMA_BF16(a, bm, acc);
        }
#pragma unroll
        for (int r = 0; r < 8; ++r)
            scores_f[(r + half * 8) * S68 + s0 + l15] = acc[r] * SCALE;
    }
    __syncthreads();

    // ---- Softmax over 64 text tokens (16 rows, 1 thread each) ----
    if (tid < 16) {
        const float* row = scores_f + tid * S68;
        float mx = row[0];
        for (int s = 1; s < TT; ++s) mx = fmaxf(mx, row[s]);
        float sum = 0.f;
        for (int s = 0; s < TT; ++s) sum += __expf(row[s] - mx);
        const float inv = 1.f / sum;
        for (int s = 0; s < TT; ++s)
            attn_bf[tid * S72 + s] = (__bf16)(__expf(row[s] - mx) * inv);
    }
    __syncthreads();

    // ---- Phase 3: ctx = attn @ v ; gate = sigmoid(ctx) ----
    {
        const int n0 = wv * 16;
        v8f acc = {};
#pragma unroll
        for (int ks = 0; ks < 2; ++ks) {
            v16bf a  = load_a_tile(attn_bf, S72, ks * 32, l15, half);
            v16bf bm = load_b_tile(vT_b, TT, n0, ks * 32, l15, half);
            acc = WMMA_BF16(a, bm, acc);
        }
        const int n = n0 + l15;
#pragma unroll
        for (int r = 0; r < 8; ++r) {
            const int m = r + half * 8;
            const float c = acc[r];
            ctx_bf[m * S136 + n]  = (__bf16)c;
            gate_bf[m * S136 + n] = (__bf16)(1.f / (1.f + __expf(-c)));
        }
    }
    __syncthreads();

    // ---- Phase 4: gamma = ctx @ Wg^T + bg ; beta = ctx @ Wb^T + bb ----
    {
        const int n0 = wv * 16;
        const float biasg = bg[n0 + l15];
        const float biasb = bb[n0 + l15];
        v8f ag, ab;
#pragma unroll
        for (int r = 0; r < 8; ++r) { ag[r] = biasg; ab[r] = biasb; }
#pragma unroll
        for (int ks = 0; ks < 4; ++ks) {
            v16bf a   = load_a_tile(ctx_bf, S136, ks * 32, l15, half);
            v16bf bmg = load_b_tile(wg_bf, BK, n0, ks * 32, l15, half);
            v16bf bmb = load_b_tile(wb_bf, BK, n0, ks * 32, l15, half);
            ag = WMMA_BF16(a, bmg, ag);
            ab = WMMA_BF16(a, bmb, ab);
        }
        store_c_bf(gamma_bf, S136, n0, ag, l15, half);
        store_c_bf(beta_bf,  S136, n0, ab, l15, half);
    }

    // Prefetch the residual rows this wave will re-read in phase 5 so the L2
    // refetch overlaps the barrier (emits global_prefetch_b8).
    {
        const int g = wv;
        __builtin_prefetch(hp_t + (size_t)(half * 8) * DEQ + g * BK + l15 * 8, 0, 3);
        __builtin_prefetch(hp_t + (size_t)(half * 8 + 4) * DEQ + g * BK + l15 * 8, 0, 3);
    }
    __syncthreads();

    // ---- Phase 5: equi_out = X_g @ We^T ; fused modulation + residual ----
    // Wave w owns group g = w. The four 16x32 A tiles are loaded ONCE into
    // registers and reused across all 8 N-tiles (8x less LDS traffic in the
    // dominant GEMM).
    {
        const int g = wv;
        const __bf16* hg = h_bf + g * BK;    // A base: rows 0..15, K = group cols
        v16bf a0 = load_a_tile(hg, HS, 0,  l15, half);
        v16bf a1 = load_a_tile(hg, HS, 32, l15, half);
        v16bf a2 = load_a_tile(hg, HS, 64, l15, half);
        v16bf a3 = load_a_tile(hg, HS, 96, l15, half);
#pragma unroll
        for (int nt = 0; nt < 8; ++nt) {
            const int n0 = nt * 16;
            v8f acc = {};
            acc = WMMA_BF16(a0, load_b_tile(we_bf, BK, n0, 0,  l15, half), acc);
            acc = WMMA_BF16(a1, load_b_tile(we_bf, BK, n0, 32, l15, half), acc);
            acc = WMMA_BF16(a2, load_b_tile(we_bf, BK, n0, 64, l15, half), acc);
            acc = WMMA_BF16(a3, load_b_tile(we_bf, BK, n0, 96, l15, half), acc);
            const int n = n0 + l15;
#pragma unroll
            for (int r = 0; r < 8; ++r) {
                const int m = r + half * 8;
                const int col = g * BK + n;
                const float hv = hp_t[(size_t)m * DEQ + col];       // residual (L2 hot)
                const float gm = (float)gamma_bf[m * S136 + n];
                const float bt = (float)beta_bf[m * S136 + n];
                const float gt = (float)gate_bf[m * S136 + n];
                op_t[(size_t)m * DEQ + col] = hv + gt * fmaf(gm, acc[r], bt);
            }
        }
    }
}

// ---------------------------------------------------------------------------
// Host launcher
// ---------------------------------------------------------------------------
extern "C" void kernel_launch(void* const* d_in, const int* in_sizes, int n_in,
                              void* d_out, int out_size, void* d_ws, size_t ws_size,
                              hipStream_t stream) {
    (void)in_sizes; (void)n_in; (void)out_size; (void)ws_size;

    const float* h_prime = (const float*)d_in[0];
    const float* h_llm   = (const float*)d_in[1];
    const float* Wq = (const float*)d_in[2];
    const float* bq = (const float*)d_in[3];
    const float* Wk = (const float*)d_in[4];
    const float* bk = (const float*)d_in[5];
    const float* Wv = (const float*)d_in[6];
    const float* bv = (const float*)d_in[7];
    const float* Wg = (const float*)d_in[8];
    const float* bg = (const float*)d_in[9];
    const float* Wb = (const float*)d_in[10];
    const float* bb = (const float*)d_in[11];
    const float* We = (const float*)d_in[12];
    float* out = (float*)d_out;

    char* ws = (char*)d_ws;
    __bf16* we_bf = (__bf16*)(ws + WS_WE);
    __bf16* wq_bf = (__bf16*)(ws + WS_WQ);
    __bf16* wg_bf = (__bf16*)(ws + WS_WG);
    __bf16* wb_bf = (__bf16*)(ws + WS_WB);
    __bf16* k_bf  = (__bf16*)(ws + WS_K);
    __bf16* vT_bf = (__bf16*)(ws + WS_VT);

    equi_prep_weights<<<64, 256, 0, stream>>>(Wq, Wg, Wb, We, wq_bf, wg_bf, wb_bf, we_bf);
    equi_kv_kernel<<<NB * TT, 128, 0, stream>>>(h_llm, Wk, bk, Wv, bv, k_bf, vT_bf);
    equi_main_kernel<<<NB * NE * (NT / 16), 256, 0, stream>>>(
        h_prime, bq, bg, bb, we_bf, wq_bf, wg_bf, wb_bf, k_bf, vT_bf, out);
}
